// Model_14336600834740
// MI455X (gfx1250) — compile-verified, matrix operand order checked by reading, not run on hardware
//
#include <hip/hip_runtime.h>

// ---------------------------------------------------------------------------
// MI455X (gfx1250) transformer forward. All matmuls run on v_wmma_f32_16x16x32_bf16
// (wave32 WMMA, f32 accumulate, bf16 operands). Weights transposed+converted to
// bf16 so B fragments load as contiguous [N][K] rows. GEMM: 128x128x64 block
// tile, double-buffered LDS filled by GLOBAL_LOAD_ASYNC_TO_LDS_B128 (ASYNCcnt),
// one barrier per 64-K step, 16 WMMAs per step. No VGPR-held prefetch tile, so
// no scratch spills.
// ---------------------------------------------------------------------------

typedef unsigned short ushort_t;
typedef unsigned int uint32_t_;
typedef __attribute__((ext_vector_type(16))) __bf16 v16bf;
typedef __attribute__((ext_vector_type(8)))  float  v8f;

#define WMMA_BF16(a, b, c) \
    __builtin_amdgcn_wmma_f32_16x16x32_bf16(false, (a), false, (b), (short)0, (c), false, false)

__device__ __forceinline__ ushort_t f2bf(float f) {
    uint32_t_ u = __float_as_uint(f);
    u = (u + 0x7FFFu + ((u >> 16) & 1u)) >> 16;   // round-to-nearest-even
    return (ushort_t)u;
}

// Async global->LDS 128-bit copy (CDNA5 GLOBAL_LOAD_ASYNC_TO_LDS_B128).
// ldsAddr: LDS byte address (low 32 bits of generic pointer, per ISA aperture
// rule). gptr: global address. Inline asm (portable
// across ROCm/amdgpu-toolchain builtin signature differences).
__device__ __forceinline__ void async_ld_b128(uint32_t_ ldsAddr, const void* gptr) {
    asm volatile("global_load_async_to_lds_b128 %0, %1, off"
                 :
                 : "v"(ldsAddr), "v"(gptr)
                 : "memory");
}

__device__ __forceinline__ void wait_async0() {
#if __has_builtin(__builtin_amdgcn_s_wait_asynccnt)
    __builtin_amdgcn_s_wait_asynccnt(0);
#else
    asm volatile("s_wait_asynccnt 0x0" ::: "memory");
#endif
}

// Load one 16x32 bf16 fragment from an LDS tile with row stride 64 (BK=64).
// ISA 7.12.2: lane l -> row = base + (l&15); K chunks (l>>4)*8 and +16 within
// the selected 32-wide K slice (ksub).
__device__ __forceinline__ v16bf ldsFrag(const ushort_t* S, int rowBase, int ksub, int lane) {
    const int row = rowBase + (lane & 15);
    const int ko  = ksub * 32 + ((lane >> 4) * 8);
    const uint4* p0 = (const uint4*)(S + row * 64 + ko);
    const uint4* p1 = (const uint4*)(S + row * 64 + ko + 16);
    uint4 a = *p0, b = *p1;
    union { uint32_t_ u[8]; v16bf v; } c;
    c.u[0] = a.x; c.u[1] = a.y; c.u[2] = a.z; c.u[3] = a.w;
    c.u[4] = b.x; c.u[5] = b.y; c.u[6] = b.z; c.u[7] = b.w;
    return c.v;
}

// ---------------------------------------------------------------------------
// Tiled WMMA GEMM: C[M,N] = A[M,K](bf16) * Bt[N,K](bf16)^T (+bias)(+gelu)(+resid)
// Block = 256 threads = 8 wave32. Wave grid 2x4, each wave computes 64x32 via
// 4x2 wmma accumulators. K must be a multiple of 64 (true for all uses here).
// ---------------------------------------------------------------------------
template <int GELU>
__global__ __launch_bounds__(256) void gemm_bf16(
    const ushort_t* __restrict__ A, long long aB, int lda,
    const ushort_t* __restrict__ Bt, long long bB, int ldb,
    const float* __restrict__ bias,
    const float* __restrict__ resid, long long rB,
    float* __restrict__ Cf, long long cfB,
    ushort_t* __restrict__ Cb, long long cbB,
    ushort_t* __restrict__ Ct, long long ctB, int ldct,
    int ldc, int M, int N, int K) {

    __shared__ ushort_t As[2][128 * 64];
    __shared__ ushort_t Bs[2][128 * 64];

    const int bz = blockIdx.z;
    const int m0 = blockIdx.y * 128;
    const int n0 = blockIdx.x * 128;

    A += (size_t)bz * aB;
    Bt += (size_t)bz * bB;
    if (Cf)    Cf += (size_t)bz * cfB;
    if (Cb)    Cb += (size_t)bz * cbB;
    if (Ct)    Ct += (size_t)bz * ctB;
    if (resid) resid += (size_t)bz * rB;

    const int tid  = threadIdx.x;
    const int lane = tid & 31;
    const int wave = tid >> 5;
    const int wm   = (wave >> 2) * 64;  // wave row origin in tile
    const int wn   = (wave & 3) * 32;   // wave col origin in tile

    // Global->LDS staging: each thread owns 32 bf16 (4x async b128) of A and of
    // Bt. Row index is CLAMPED (not predicated): out-of-range rows duplicate
    // the last valid row into the LDS tile; their outputs are masked in the
    // epilogue, so no exec-divergent load path is generated.
    const int lr = tid >> 1;
    const int lc = (tid & 1) * 32;
    const int arow = (m0 + lr < M) ? (m0 + lr) : (M - 1);
    const int brow = (n0 + lr < N) ? (n0 + lr) : (N - 1);
    const ushort_t* Ap = A + (size_t)arow * lda + lc;
    const ushort_t* Bp = Bt + (size_t)brow * ldb + lc;

    const uint32_t_ laBase[2] = {
        (uint32_t_)(unsigned long)(&As[0][lr * 64 + lc]),
        (uint32_t_)(unsigned long)(&As[1][lr * 64 + lc])};
    const uint32_t_ lbBase[2] = {
        (uint32_t_)(unsigned long)(&Bs[0][lr * 64 + lc]),
        (uint32_t_)(unsigned long)(&Bs[1][lr * 64 + lc])};

#define ISSUE_TILE(buf, kk)                                      \
    do {                                                         \
        const ushort_t* ap_ = Ap + (kk);                         \
        const ushort_t* bp_ = Bp + (kk);                         \
        const uint32_t_ la_ = laBase[buf];                       \
        const uint32_t_ lb_ = lbBase[buf];                       \
        async_ld_b128(la_,      ap_);                            \
        async_ld_b128(la_ + 16, ap_ + 8);                        \
        async_ld_b128(la_ + 32, ap_ + 16);                       \
        async_ld_b128(la_ + 48, ap_ + 24);                       \
        async_ld_b128(lb_,      bp_);                            \
        async_ld_b128(lb_ + 16, bp_ + 8);                        \
        async_ld_b128(lb_ + 32, bp_ + 16);                       \
        async_ld_b128(lb_ + 48, bp_ + 24);                       \
    } while (0)

    v8f acc[4][2];
#pragma unroll
    for (int i = 0; i < 4; i++)
#pragma unroll
        for (int j = 0; j < 2; j++) acc[i][j] = (v8f)(0.0f);

    ISSUE_TILE(0, 0);
    wait_async0();
    __syncthreads();

    int pb = 0;
    for (int k = 0; k < K; k += 64) {
        const bool more = (k + 64) < K;
        // DMA next tile into the other buffer; overlaps with the WMMAs below.
        if (more) ISSUE_TILE(pb ^ 1, k + 64);

#pragma unroll
        for (int ks = 0; ks < 2; ks++) {
            v16bf af[4], bfx[2];
#pragma unroll
            for (int fm = 0; fm < 4; fm++) af[fm] = ldsFrag(As[pb], wm + fm * 16, ks, lane);
#pragma unroll
            for (int fn = 0; fn < 2; fn++) bfx[fn] = ldsFrag(Bs[pb], wn + fn * 16, ks, lane);
#pragma unroll
            for (int fm = 0; fm < 4; fm++)
#pragma unroll
                for (int fn = 0; fn < 2; fn++)
                    acc[fm][fn] = WMMA_BF16(af[fm], bfx[fn], acc[fm][fn]);
        }

        if (more) wait_async0();  // own transfers done; barrier publishes to WG
        __syncthreads();
        pb ^= 1;
    }
#undef ISSUE_TILE

    // Epilogue. C layout (ISA): v8f elem i -> m = base + (lane<16 ? i : i+8),
    // n = base + (lane&15).
#pragma unroll
    for (int fm = 0; fm < 4; fm++) {
#pragma unroll
        for (int fn = 0; fn < 2; fn++) {
            const int nb = n0 + wn + fn * 16 + (lane & 15);
            const float bv = (bias && nb < N) ? bias[nb] : 0.0f;
            const int mrb = m0 + wm + fm * 16 + ((lane >> 4) ? 8 : 0);
#pragma unroll
            for (int i = 0; i < 8; i++) {
                const int m = mrb + i;
                if (m < M && nb < N) {
                    float v = acc[fm][fn][i] + bv;
                    if (GELU) v = 0.5f * v * (1.0f + erff(v * 0.70710678118654752f));
                    if (resid) v += resid[(size_t)m * ldc + nb];
                    if (Cf) Cf[(size_t)m * ldc + nb] = v;
                    if (Cb) Cb[(size_t)m * ldc + nb] = f2bf(v);
                    if (Ct) Ct[(size_t)nb * ldct + m] = f2bf(v);
                }
            }
        }
    }
}

// ---------------------------------------------------------------------------
// Weight transpose + f32->bf16 convert: in [z][R][C] f32 -> out [z][C][R] bf16
// ---------------------------------------------------------------------------
__global__ __launch_bounds__(256) void transpose_cvt(const float* __restrict__ in,
                                                     ushort_t* __restrict__ out,
                                                     int R, int C) {
    __shared__ float t[32][33];
    const size_t base = (size_t)blockIdx.z * R * C;
    const int c0 = blockIdx.x * 32, r0 = blockIdx.y * 32;
    for (int i = threadIdx.y; i < 32; i += 8) {
        int r = r0 + i, c = c0 + threadIdx.x;
        t[i][threadIdx.x] = (r < R && c < C) ? in[base + (size_t)r * C + c] : 0.0f;
    }
    __syncthreads();
    for (int i = threadIdx.y; i < 32; i += 8) {
        int c = c0 + i, r = r0 + threadIdx.x;
        if (c < C && r < R) out[base + (size_t)c * R + r] = f2bf(t[threadIdx.x][i]);
    }
}

// ---------------------------------------------------------------------------
// Point encoder features: [sin(x0*f), cos(x0*f), sin(x1*f), cos(x1*f)] -> bf16
// ---------------------------------------------------------------------------
__global__ __launch_bounds__(256) void point_enc(const float* __restrict__ x,
                                                 const float* __restrict__ freqs,
                                                 ushort_t* __restrict__ feat, int Np) {
    const int n = blockIdx.x, b = blockIdx.y;
    const float x0 = x[((size_t)b * Np + n) * 2 + 0];
    const float x1 = x[((size_t)b * Np + n) * 2 + 1];
    ushort_t* out = feat + ((size_t)b * Np + n) * 2048;
    for (int j = threadIdx.x; j < 512; j += 256) {
        const float f = freqs[j];
        const float a0 = x0 * f, a1 = x1 * f;
        out[j]        = f2bf(sinf(a0));
        out[512 + j]  = f2bf(cosf(a0));
        out[1024 + j] = f2bf(sinf(a1));
        out[1536 + j] = f2bf(cosf(a1));
    }
}

__global__ __launch_bounds__(256) void write_cls(const float* __restrict__ cls,
                                                 float* __restrict__ hf,
                                                 ushort_t* __restrict__ hb,
                                                 int S, int D) {
    const int b = blockIdx.x;
    for (int d = threadIdx.x; d < D; d += 256) {
        const float v = cls[d];
        hf[(size_t)b * S * D + d] = v;
        hb[(size_t)b * S * D + d] = f2bf(v);
    }
}

// softmax over last axis with logit scale folded in; writes bf16 probs
__global__ __launch_bounds__(256) void softmax_rows(const float* __restrict__ S_,
                                                    ushort_t* __restrict__ P,
                                                    int n, float inv) {
    __shared__ float red[256];
    const size_t base = ((size_t)blockIdx.y * gridDim.x + blockIdx.x) * n;
    const int tid = threadIdx.x;
    float v[4];
    float lmax = -3.4e38f;
#pragma unroll
    for (int i = 0; i < 4; i++) {
        v[i] = S_[base + tid + i * 256] * inv;
        lmax = fmaxf(lmax, v[i]);
    }
    red[tid] = lmax; __syncthreads();
    for (int st = 128; st > 0; st >>= 1) {
        if (tid < st) red[tid] = fmaxf(red[tid], red[tid + st]);
        __syncthreads();
    }
    const float m = red[0]; __syncthreads();
    float ls = 0.0f;
#pragma unroll
    for (int i = 0; i < 4; i++) { v[i] = __expf(v[i] - m); ls += v[i]; }
    red[tid] = ls; __syncthreads();
    for (int st = 128; st > 0; st >>= 1) {
        if (tid < st) red[tid] += red[tid + st];
        __syncthreads();
    }
    const float rs = 1.0f / red[0]; __syncthreads();
#pragma unroll
    for (int i = 0; i < 4; i++) P[base + tid + i * 256] = f2bf(v[i] * rs);
}

// h2 = LayerNorm(att + h) * g + beta; writes f32 + bf16
__global__ __launch_bounds__(256) void ln_resid(const float* __restrict__ att,
                                                const float* __restrict__ hin,
                                                const float* __restrict__ g,
                                                const float* __restrict__ beta,
                                                float* __restrict__ h2f,
                                                ushort_t* __restrict__ h2b, int D) {
    __shared__ float red[256];
    const size_t base = ((size_t)blockIdx.y * gridDim.x + blockIdx.x) * D;
    const int tid = threadIdx.x;
    float v[4];
    float ls = 0.0f;
#pragma unroll
    for (int i = 0; i < 4; i++) {
        const int d = tid + i * 256;
        v[i] = att[base + d] + hin[base + d];
        ls += v[i];
    }
    red[tid] = ls; __syncthreads();
    for (int st = 128; st > 0; st >>= 1) {
        if (tid < st) red[tid] += red[tid + st];
        __syncthreads();
    }
    const float mu = red[0] / (float)D; __syncthreads();
    float lv = 0.0f;
#pragma unroll
    for (int i = 0; i < 4; i++) { const float d0 = v[i] - mu; lv += d0 * d0; }
    red[tid] = lv; __syncthreads();
    for (int st = 128; st > 0; st >>= 1) {
        if (tid < st) red[tid] += red[tid + st];
        __syncthreads();
    }
    const float rstd = rsqrtf(red[0] / (float)D + 1e-5f); __syncthreads();
#pragma unroll
    for (int i = 0; i < 4; i++) {
        const int d = tid + i * 256;
        const float o = (v[i] - mu) * rstd * g[d] + beta[d];
        h2f[base + d] = o;
        h2b[base + d] = f2bf(o);
    }
}

// out[b,:] = h[b,0,:] @ Wf + bf
__global__ __launch_bounds__(256) void final_head(const float* __restrict__ h,
                                                  const float* __restrict__ Wf,
                                                  const float* __restrict__ bfv,
                                                  float* __restrict__ out, int S, int D) {
    __shared__ float red[256];
    const int b = blockIdx.x;
    const float* hr = h + (size_t)b * S * D;
    float part[10];
#pragma unroll
    for (int c = 0; c < 10; c++) part[c] = 0.0f;
    for (int d = threadIdx.x; d < D; d += 256) {
        const float hv = hr[d];
#pragma unroll
        for (int c = 0; c < 10; c++) part[c] += hv * Wf[d * 10 + c];
    }
    for (int c = 0; c < 10; c++) {
        red[threadIdx.x] = part[c]; __syncthreads();
        for (int st = 128; st > 0; st >>= 1) {
            if (threadIdx.x < st) red[threadIdx.x] += red[threadIdx.x + st];
            __syncthreads();
        }
        if (threadIdx.x == 0) out[b * 10 + c] = red[0] + bfv[c];
        __syncthreads();
    }
}

// ---------------------------------------------------------------------------
// Host-side orchestration
// ---------------------------------------------------------------------------
extern "C" void kernel_launch(void* const* d_in, const int* in_sizes, int n_in,
                              void* d_out, int out_size, void* d_ws, size_t ws_size,
                              hipStream_t stream) {
    (void)in_sizes; (void)n_in; (void)out_size; (void)ws_size;
    const int B = 16, S = 1024, D = 1024, Np = 1023, L = 8, H = 512, F = 4096;

    const float* x     = (const float*)d_in[0];
    const float* freqs = (const float*)d_in[1];
    const float* wenc  = (const float*)d_in[2];
    const float* cls   = (const float*)d_in[3];
    const float* Wq    = (const float*)d_in[4];
    const float* bq    = (const float*)d_in[5];
    const float* Wk    = (const float*)d_in[6];
    const float* bk    = (const float*)d_in[7];
    const float* Wv    = (const float*)d_in[8];
    const float* bv    = (const float*)d_in[9];
    const float* We    = (const float*)d_in[10];
    const float* be    = (const float*)d_in[11];
    const float* Wo    = (const float*)d_in[12];
    const float* bo    = (const float*)d_in[13];
    const float* lng   = (const float*)d_in[14];
    const float* lnb   = (const float*)d_in[15];
    const float* Wf    = (const float*)d_in[16];
    const float* bff   = (const float*)d_in[17];
    float* out = (float*)d_out;

    // workspace carve (256B aligned)
    char* ws = (char*)d_ws;
    size_t off = 0;
    auto carve = [&](size_t bytes) -> char* {
        char* p = ws + off;
        off = (off + bytes + 255) & ~(size_t)255;
        return p;
    };
    ushort_t* wenct = (ushort_t*)carve((size_t)2048 * 1024 * 2);
    ushort_t* wqt   = (ushort_t*)carve((size_t)L * H * D * 2);
    ushort_t* wkt   = (ushort_t*)carve((size_t)L * H * D * 2);
    ushort_t* wvt   = (ushort_t*)carve((size_t)L * D * D * 2);
    ushort_t* wet   = (ushort_t*)carve((size_t)L * F * D * 2);
    ushort_t* wot   = (ushort_t*)carve((size_t)L * D * F * 2);
    float*    h_f   = (float*)carve((size_t)B * S * D * 4);
    ushort_t* h_b   = (ushort_t*)carve((size_t)B * S * D * 2);
    float*    h2_f  = (float*)carve((size_t)B * S * D * 4);
    ushort_t* h2_b  = (ushort_t*)carve((size_t)B * S * D * 2);
    ushort_t* qh    = (ushort_t*)carve((size_t)B * S * H * 2);
    ushort_t* kh    = (ushort_t*)carve((size_t)B * S * H * 2);
    ushort_t* vt    = (ushort_t*)carve((size_t)B * D * S * 2);
    float*    sc    = (float*)carve((size_t)B * S * S * 4);   // scores; att aliases
    ushort_t* Pb    = (ushort_t*)carve((size_t)B * S * S * 2);
    ushort_t* eb    = (ushort_t*)carve((size_t)B * S * F * 2); // gelu act; feat aliases
    float*    att   = sc;                                      // alias (scores dead)
    ushort_t* feat  = eb;                                      // alias (feat dead by layer 0)

    const dim3 blk(256);

    // 1) weights -> bf16, transposed to [N][K]
    transpose_cvt<<<dim3(1024 / 32, 2048 / 32, 1), dim3(32, 8), 0, stream>>>(wenc, wenct, 2048, 1024);
    transpose_cvt<<<dim3(H / 32, D / 32, L), dim3(32, 8), 0, stream>>>(Wq, wqt, D, H);
    transpose_cvt<<<dim3(H / 32, D / 32, L), dim3(32, 8), 0, stream>>>(Wk, wkt, D, H);
    transpose_cvt<<<dim3(D / 32, D / 32, L), dim3(32, 8), 0, stream>>>(Wv, wvt, D, D);
    transpose_cvt<<<dim3(F / 32, D / 32, L), dim3(32, 8), 0, stream>>>(We, wet, D, F);
    transpose_cvt<<<dim3(D / 32, F / 32, L), dim3(32, 8), 0, stream>>>(Wo, wot, F, D);

    // 2) point-encoder features (bf16)
    point_enc<<<dim3(Np, B), blk, 0, stream>>>(x, freqs, feat, Np);

    // 3) h[:,1:,:] = feat @ w_enc ; h[:,0,:] = cls
    gemm_bf16<0><<<dim3(8, 8, B), blk, 0, stream>>>(
        feat, (long long)Np * 2048, 2048, wenct, 0, 2048, nullptr, nullptr, 0,
        h_f + D, (long long)S * D, h_b + D, (long long)S * D, nullptr, 0, 0,
        D, Np, D, 2048);
    write_cls<<<dim3(B), blk, 0, stream>>>(cls, h_f, h_b, S, D);

    // 4) transformer layers
    for (int l = 0; l < L; l++) {
        // qh = h @ Wq + bq  (bf16 out)
        gemm_bf16<0><<<dim3(4, 8, B), blk, 0, stream>>>(
            h_b, (long long)S * D, D, wqt + (size_t)l * H * D, 0, D, bq + (size_t)l * H,
            nullptr, 0, nullptr, 0, qh, (long long)S * H, nullptr, 0, 0,
            H, S, H, D);
        // kh = h @ Wk + bk  (bf16 out)
        gemm_bf16<0><<<dim3(4, 8, B), blk, 0, stream>>>(
            h_b, (long long)S * D, D, wkt + (size_t)l * H * D, 0, D, bk + (size_t)l * H,
            nullptr, 0, nullptr, 0, kh, (long long)S * H, nullptr, 0, 0,
            H, S, H, D);
        // vt = (h @ Wv + bv)^T  (bf16, transposed epilogue -> [D][S])
        gemm_bf16<0><<<dim3(8, 8, B), blk, 0, stream>>>(
            h_b, (long long)S * D, D, wvt + (size_t)l * D * D, 0, D, bv + (size_t)l * D,
            nullptr, 0, nullptr, 0, nullptr, 0, vt, (long long)D * S, S,
            D, S, D, D);
        // scores[q,k] = qh @ kh^T   (kh already row-major [k][d] = Bt)
        gemm_bf16<0><<<dim3(8, 8, B), blk, 0, stream>>>(
            qh, (long long)S * H, H, kh, (long long)S * H, H, nullptr, nullptr, 0,
            sc, (long long)S * S, nullptr, 0, nullptr, 0, 0,
            S, S, S, H);
        // P = softmax_k(scores / 16)  (bf16 out)
        softmax_rows<<<dim3(S, B), blk, 0, stream>>>(sc, Pb, S, 1.0f / 16.0f);
        // att = P @ V   (Bt = Vt [d][s])
        gemm_bf16<0><<<dim3(8, 8, B), blk, 0, stream>>>(
            Pb, (long long)S * S, S, vt, (long long)D * S, S, nullptr, nullptr, 0,
            att, (long long)S * D, nullptr, 0, nullptr, 0, 0,
            D, S, D, S);
        // h2 = LN(att + h)
        ln_resid<<<dim3(S, B), blk, 0, stream>>>(att, h_f, lng + (size_t)l * D,
                                                 lnb + (size_t)l * D, h2_f, h2_b, D);
        // e = gelu(h2 @ We + be)  (bf16 out)
        gemm_bf16<1><<<dim3(32, 8, B), blk, 0, stream>>>(
            h2_b, (long long)S * D, D, wet + (size_t)l * F * D, 0, D, be + (size_t)l * F,
            nullptr, 0, nullptr, 0, eb, (long long)S * F, nullptr, 0, 0,
            F, S, F, D);
        // h = e @ Wo + bo + h2  (f32 + bf16 out)
        gemm_bf16<0><<<dim3(8, 8, B), blk, 0, stream>>>(
            eb, (long long)S * F, F, wot + (size_t)l * D * F, 0, F, bo + (size_t)l * D,
            h2_f, (long long)S * D,
            h_f, (long long)S * D, h_b, (long long)S * D, nullptr, 0, 0,
            D, S, D, F);
    }

    // 5) out = (h[:,0,:] @ Wf + bf)
    final_head<<<dim3(B), blk, 0, stream>>>(h_f, Wf, bff, out, S, D);
}
